// CausalSelfAttention_29205777612958
// MI455X (gfx1250) — matrix operand order repaired and checked by reading, not compiled, not run on hardware
//
#include <hip/hip_runtime.h>

#define NH   16
#define NKV  4
#define HD   128
#define DIM  2048
#define KVD  512
#define SEQ  2048
#define KC   128          // K-chunk staged in LDS per TDM transfer

typedef __attribute__((ext_vector_type(16))) __bf16         v16bf;
typedef __attribute__((ext_vector_type(8)))  float          v8f;
typedef __attribute__((ext_vector_type(8)))  unsigned short v8u;
typedef __attribute__((ext_vector_type(4)))  unsigned int   v4ui;
typedef __attribute__((ext_vector_type(8)))  int            v8i;
typedef __attribute__((ext_vector_type(4)))  int            v4i;

#if __has_builtin(__builtin_amdgcn_tensor_load_to_lds) && __has_builtin(__builtin_amdgcn_s_wait_tensorcnt)
#define HAVE_TDM 1
#else
#define HAVE_TDM 0
#endif

union FragU {
  struct { v8u lo, hi; } p;
  v16bf v;
};

__device__ __forceinline__ unsigned short f2bf(float f) {
  unsigned int u = __float_as_uint(f);
  u += 0x7FFFu + ((u >> 16) & 1u);   // round-to-nearest-even
  return (unsigned short)(u >> 16);
}

// A-matrix fragment: lane holds row (l&15); two 8-half chunks at K = kB and 8+kB
__device__ __forceinline__ v16bf ldA(const unsigned short* p0, const unsigned short* p1) {
  FragU f; f.p.lo = *(const v8u*)p0; f.p.hi = *(const v8u*)(p1); return f.v;
}
// B-matrix fragment: lane holds col (l&15); 16 consecutive K halfs at K = 16*(l>=16)
__device__ __forceinline__ v16bf ldB(const unsigned short* p) {
  FragU f; f.p.lo = *(const v8u*)p; f.p.hi = *(const v8u*)(p + 8); return f.v;
}

__device__ __forceinline__ v8f wmma_bf16(v16bf a, v16bf b, v8f c) {
  return __builtin_amdgcn_wmma_f32_16x16x32_bf16(false, a, false, b, (short)0, c, false, false);
}

#if HAVE_TDM
// Issue one TDM 2D tile load: rows x cols (bf16 halfs) from row-major global (ld = ld_halfs)
// into contiguous LDS at byte offset lds_off. D# per CDNA5 ISA ch.8 (group0 + group1; 2D).
// amdgpu-toolchain (clang-23) 6-arg builtin: (v4u g0, v8i g1, v4i, v4i, v8i, i32 cpol).
__device__ __forceinline__ void tdm_load_2d(unsigned lds_off, const unsigned short* gptr,
                                            unsigned rows, unsigned cols, unsigned ld_halfs) {
  unsigned long long ga = (unsigned long long)gptr;
  v4ui g0;
  g0[0] = 1u;                                             // count=1, user descriptor
  g0[1] = lds_off;                                        // LDS byte address
  g0[2] = (unsigned)(ga & 0xFFFFFFFFu);                   // global_addr[31:0]
  g0[3] = (unsigned)((ga >> 32) & 0x01FFFFFFu) | (2u << 30); // addr[56:32] | type=2
  v8i g1;
  g1[0] = (int)(1u << 16);                                // data_size=1 (2 bytes); mask=0
  g1[1] = (int)((cols & 0xFFFFu) << 16);                  // tensor_dim0[15:0]
  g1[2] = (int)(((cols >> 16) & 0xFFFFu) | ((rows & 0xFFFFu) << 16)); // td0 hi | td1 lo
  g1[3] = (int)(((rows >> 16) & 0xFFFFu) | (cols << 16)); // td1 hi | tile_dim0
  g1[4] = (int)rows;                                      // tile_dim1 | tile_dim2=0
  g1[5] = (int)ld_halfs;                                  // tensor_dim0_stride[31:0]
  g1[6] = 0;                                              // stride0 hi | stride1 lo
  g1[7] = 0;
  v4i z4 = {0, 0, 0, 0};
  v8i z8 = {0, 0, 0, 0, 0, 0, 0, 0};
  __builtin_amdgcn_tensor_load_to_lds(g0, g1, z4, z4, z8, 0);
}
#endif

// ---------------------------------------------------------------- fp32 -> bf16 pack
__global__ void pack_bf16(const float* __restrict__ src, unsigned short* __restrict__ dst, int n) {
  for (int i = blockIdx.x * blockDim.x + threadIdx.x; i < n; i += gridDim.x * blockDim.x)
    dst[i] = f2bf(src[i]);
}

// V fp32 (B,S,HKV*D) -> bf16 transposed (B,HKV,D,S) so attention B-frags are contiguous
__global__ void pack_v_t(const float* __restrict__ vf, unsigned short* __restrict__ vb, int Bn) {
  int n = Bn * SEQ * NKV * HD;
  for (int i = blockIdx.x * blockDim.x + threadIdx.x; i < n; i += gridDim.x * blockDim.x) {
    int d  = i % HD;
    int hv = (i / HD) % NKV;
    int s  = (i / (HD * NKV)) % SEQ;
    int b  = i / (HD * NKV * SEQ);
    vb[(((size_t)b * NKV + hv) * HD + d) * SEQ + s] = f2bf(vf[i]);
  }
}

// ---------------------------------------------------------------- GEMM  C = A * B^T
// A: MxK bf16 row-major, B: NxK bf16 row-major, C: MxN fp32.
// Block = 4 waves, output 32x256. B panel (256 x KC) staged in LDS by the Tensor Data
// Mover, double-buffered; TDM chunk c+1 overlaps the 32 WMMAs/wave of chunk c.
__global__ __launch_bounds__(128) void gemm_bf16_nt_tdm(
    const unsigned short* __restrict__ A, const unsigned short* __restrict__ B,
    float* __restrict__ C, int K, int lda, int ldb, int ldc) {
  __shared__ __align__(16) unsigned short smB[2][256 * KC];   // 2 x 64 KB
  const int lane = threadIdx.x & 31;
  const int wave = threadIdx.x >> 5;
  const int r    = lane & 15;
  const int hi   = lane >> 4;
  const int koff = hi * 8;
  const int kB   = hi * 16;
  const int m0   = blockIdx.y * 32;
  const int nblk = blockIdx.x * 256;
  v8f acc[2][4] = {};
  const unsigned short* arow0 = A + (size_t)(m0 + r) * lda;
  const unsigned short* arow1 = A + (size_t)(m0 + 16 + r) * lda;
  const unsigned short* bpan  = B + (size_t)nblk * ldb;
  const int nchunk = K / KC;

#if HAVE_TDM
  const unsigned lds0 = (unsigned)(size_t)(void*)&smB[0][0];
  const unsigned lds1 = (unsigned)(size_t)(void*)&smB[1][0];
  if (wave == 0) tdm_load_2d(lds0, bpan, 256, KC, (unsigned)ldb);
#endif

  for (int c = 0; c < nchunk; ++c) {
    const int k0g = c * KC;
#if HAVE_TDM
    if (wave == 0) __builtin_amdgcn_s_wait_tensorcnt(0);  // chunk c landed in LDS
    __syncthreads();                                      // publish to all 4 waves
    if (wave == 0 && c + 1 < nchunk)                      // prefetch next panel chunk
      tdm_load_2d(((c + 1) & 1) ? lds1 : lds0, bpan + k0g + KC, 256, KC, (unsigned)ldb);
#else
    for (int idx = threadIdx.x; idx < 256 * (KC / 8); idx += 128) {
      int row = idx / (KC / 8), c8 = idx % (KC / 8);
      *(v8u*)&smB[c & 1][row * KC + c8 * 8] =
          *(const v8u*)(bpan + (size_t)row * ldb + k0g + c8 * 8);
    }
    __syncthreads();
#endif
    __builtin_prefetch(arow0 + k0g + KC, 0, 0);           // global_prefetch_b8 on A stream
    const unsigned short* bs = &smB[c & 1][0];
#pragma unroll
    for (int kk = 0; kk < KC; kk += 32) {
      v16bf a0 = ldA(arow0 + k0g + kk + kB, arow0 + k0g + kk + 8 + kB);
      v16bf a1 = ldA(arow1 + k0g + kk + kB, arow1 + k0g + kk + 8 + kB);
#pragma unroll
      for (int t = 0; t < 4; ++t) {
        v16bf bf = ldB(bs + (wave * 64 + t * 16 + r) * KC + kk + kB);
        acc[0][t] = wmma_bf16(a0, bf, acc[0][t]);
        acc[1][t] = wmma_bf16(a1, bf, acc[1][t]);
      }
    }
#if !HAVE_TDM
    __syncthreads();
#endif
  }
#pragma unroll
  for (int mt = 0; mt < 2; ++mt)
#pragma unroll
    for (int t = 0; t < 4; ++t)
#pragma unroll
      for (int v = 0; v < 8; ++v)
        C[(size_t)(m0 + mt * 16 + v + hi * 8) * ldc + nblk + wave * 64 + t * 16 + r] =
            acc[mt][t][v];
}

// ---------------------------------------------------------------- RMSNorm + RoPE + gain/scale
// src: (rows = B*S*heads) x 128 fp32 ; dst: (B, heads, S, 128) bf16. One wave per row.
__global__ __launch_bounds__(32) void rmsnorm_rope(
    const float* __restrict__ src, unsigned short* __restrict__ dst,
    const float* __restrict__ gain, int nheads, float extra_scale) {
  const int rid = blockIdx.x;
  const int h = rid % nheads;
  const int s = (rid / nheads) % SEQ;
  const int b = rid / (nheads * SEQ);
  const float* x = src + (size_t)rid * HD;
  const int t = threadIdx.x;
  float x0 = x[t], x1 = x[t + 32], x2 = x[t + 64], x3 = x[t + 96];
  float ss = x0 * x0 + x1 * x1 + x2 * x2 + x3 * x3;
#pragma unroll
  for (int m = 1; m < 32; m <<= 1) ss += __shfl_xor(ss, m, 32);
  const float rn = __frsqrt_rn(ss * (1.0f / 128.0f) + 1.1920929e-07f);
  const float g = (gain ? gain[h] : 1.0f) * extra_scale;
  const float n0 = x0 * rn, n1 = x1 * rn, n2 = x2 * rn, n3 = x3 * rn;
  const float pos = (float)s;
  const float f0 = __powf(10000.0f, -(float)(2 * t) / 128.0f);
  const float f1 = __powf(10000.0f, -(float)(2 * (t + 32)) / 128.0f);
  float s0, c0, s1, c1;
  __sincosf(pos * f0, &s0, &c0);
  __sincosf(pos * f1, &s1, &c1);
  unsigned short* o = dst + (((size_t)b * nheads + h) * SEQ + s) * HD;
  o[t]      = f2bf(g * ( n0 * c0 + n2 * s0));
  o[t + 64] = f2bf(g * (-n0 * s0 + n2 * c0));
  o[t + 32] = f2bf(g * ( n1 * c1 + n3 * s1));
  o[t + 96] = f2bf(g * (-n1 * s1 + n3 * c1));
}

// ---------------------------------------------------------------- Flash attention (causal, GQA)
// Q:(B,H,S,D) K:(B,HKV,S,D) Vt:(B,HKV,D,S) bf16 -> Y:(B,S,H*D) fp32. One wave per 16-query tile.
#define NEGINF -3.0e38f
__global__ __launch_bounds__(128) void flash_attn_wmma(
    const unsigned short* __restrict__ Q, const unsigned short* __restrict__ Kt,
    const unsigned short* __restrict__ Vt, float* __restrict__ Y) {
  __shared__ __align__(16) unsigned short Plds[4][16 * 32];
  const int lane = threadIdx.x & 31;
  const int wave = threadIdx.x >> 5;
  const int r    = lane & 15;
  const int hi   = lane >> 4;
  const int koff = hi * 8;
  const int kB   = hi * 16;
  const int h  = blockIdx.y;
  const int b  = blockIdx.z;
  const int hv = h >> 2;                       // g = H/HKV = 4
  const int m0 = (blockIdx.x * 4 + wave) * 16;
  const unsigned short* Qh = Q + (((size_t)b * NH + h) * SEQ + m0 + r) * HD;
  const unsigned short* Kh = Kt + (((size_t)b * NKV + hv) * SEQ) * HD;
  const unsigned short* Vh = Vt + (((size_t)b * NKV + hv) * HD) * (size_t)SEQ;
  unsigned short* pl = &Plds[wave][0];

  v16bf qf[4];
#pragma unroll
  for (int t = 0; t < 4; ++t)
    qf[t] = ldA(Qh + t * 32 + kB, Qh + t * 32 + 8 + kB);

  v8f o[8] = {};
  float mr[8], lr[8];
#pragma unroll
  for (int v = 0; v < 8; ++v) { mr[v] = NEGINF; lr[v] = 0.0f; }

  const int jend = m0 + 16;
  for (int j0 = 0; j0 < jend; j0 += 32) {
    // ---- scores: two 16x16 tiles over 32 keys (Q already carries gain/sqrt(D))
    v8f sc0 = {}, sc1 = {};
    int jr0 = j0 + r;       if (jr0 > SEQ - 1) jr0 = SEQ - 1;
    int jr1 = j0 + 16 + r;  if (jr1 > SEQ - 1) jr1 = SEQ - 1;
    const unsigned short* kp0 = Kh + (size_t)jr0 * HD + kB;
    const unsigned short* kp1 = Kh + (size_t)jr1 * HD + kB;
#pragma unroll
    for (int t = 0; t < 4; ++t) {
      sc0 = wmma_bf16(qf[t], ldB(kp0 + t * 32), sc0);
      sc1 = wmma_bf16(qf[t], ldB(kp1 + t * 32), sc1);
    }
    // ---- online softmax; rows of the C-layout live in one 16-lane half
    const int c0 = j0 + r, c1 = j0 + 16 + r;
#pragma unroll
    for (int v = 0; v < 8; ++v) {
      const int row = m0 + v + hi * 8;
      float s0 = (c0 <= row) ? sc0[v] : NEGINF;
      float s1 = (c1 <= row) ? sc1[v] : NEGINF;
      float mt = fmaxf(s0, s1);
      mt = fmaxf(mt, __shfl_xor(mt, 1, 32));
      mt = fmaxf(mt, __shfl_xor(mt, 2, 32));
      mt = fmaxf(mt, __shfl_xor(mt, 4, 32));
      mt = fmaxf(mt, __shfl_xor(mt, 8, 32));
      const float mnew = fmaxf(mr[v], mt);
      const float p0 = __expf(s0 - mnew);
      const float p1 = __expf(s1 - mnew);
      const float corr = __expf(mr[v] - mnew);
      float ps = p0 + p1;
      ps += __shfl_xor(ps, 1, 32);
      ps += __shfl_xor(ps, 2, 32);
      ps += __shfl_xor(ps, 4, 32);
      ps += __shfl_xor(ps, 8, 32);
      lr[v] = lr[v] * corr + ps;
      mr[v] = mnew;
#pragma unroll
      for (int t = 0; t < 8; ++t) o[t][v] *= corr;
      pl[(v + hi * 8) * 32 + r]      = f2bf(p0);   // C-layout -> row-major P in LDS
      pl[(v + hi * 8) * 32 + 16 + r] = f2bf(p1);
    }
    __threadfence_block();   // order LDS P stores before the A-frag reloads (dscnt wait)
    // ---- O += P(16x32) * V(32x128): P back as A-frag, V^T rows as B-frags
    v16bf pf = ldA(pl + r * 32 + kB, pl + r * 32 + 8 + kB);
    int jc = j0 + kB; if (jc > SEQ - 16) jc = SEQ - 16;   // masked cols have p=0
#pragma unroll
    for (int t = 0; t < 8; ++t) {
      const unsigned short* vp = Vh + (size_t)(t * 16 + r) * SEQ + jc;
      o[t] = wmma_bf16(pf, ldB(vp), o[t]);
    }
  }
  // ---- epilogue: normalize and scatter to (B,S,H*D)
#pragma unroll
  for (int v = 0; v < 8; ++v) lr[v] = 1.0f / lr[v];
#pragma unroll
  for (int t = 0; t < 8; ++t)
#pragma unroll
    for (int v = 0; v < 8; ++v)
      Y[((size_t)b * SEQ + m0 + v + hi * 8) * DIM + h * HD + t * 16 + r] = o[t][v] * lr[v];
}

// ---------------------------------------------------------------- host-side orchestration
extern "C" void kernel_launch(void* const* d_in, const int* in_sizes, int n_in,
                              void* d_out, int out_size, void* d_ws, size_t ws_size,
                              hipStream_t stream) {
  (void)n_in; (void)out_size; (void)ws_size;
  const float* x  = (const float*)d_in[0];
  const float* wq = (const float*)d_in[1];
  const float* wk = (const float*)d_in[2];
  const float* wv = (const float*)d_in[3];
  const float* wo = (const float*)d_in[4];
  const float* qg = (const float*)d_in[5];
  float* out = (float*)d_out;

  const int Bn = in_sizes[0] / (SEQ * DIM);   // = 2
  const int M  = Bn * SEQ;                    // = 4096

  char* ws = (char*)d_ws;
  unsigned short* xb  = (unsigned short*)(ws + 0);                       // 16.78 MB
  unsigned short* wqb = (unsigned short*)(ws + 16777216);                //  8.39 MB
  unsigned short* wkb = (unsigned short*)(ws + 25165824);                //  2.10 MB
  unsigned short* wvb = (unsigned short*)(ws + 27262976);                //  2.10 MB
  unsigned short* wob = (unsigned short*)(ws + 29360128);                //  8.39 MB
  float*          qf  = (float*)         (ws + 37748736);                // 33.55 MB
  float*          kf  = (float*)         (ws + 71303168);                //  8.39 MB
  float*          vf  = (float*)         (ws + 79691776);                //  8.39 MB
  unsigned short* qb  = (unsigned short*)(ws + 88080384);                // 16.78 MB
  unsigned short* kb  = (unsigned short*)(ws + 104857600);               //  4.19 MB
  unsigned short* vb  = (unsigned short*)(ws + 109051904);               //  4.19 MB
  float*          yf  = qf;   // qf dead after qb is built
  unsigned short* yb  = xb;   // xb dead after QKV GEMMs

  const int PB = 256, PG = 2048;
  // 1) pack inputs to bf16
  pack_bf16<<<PG, PB, 0, stream>>>(x,  xb,  M * DIM);
  pack_bf16<<<PG, PB, 0, stream>>>(wq, wqb, DIM * DIM);
  pack_bf16<<<PG, PB, 0, stream>>>(wk, wkb, KVD * DIM);
  pack_bf16<<<PG, PB, 0, stream>>>(wv, wvb, KVD * DIM);
  pack_bf16<<<PG, PB, 0, stream>>>(wo, wob, DIM * DIM);
  // 2) QKV projections (TDM-staged bf16 WMMA, fp32 accumulate)
  gemm_bf16_nt_tdm<<<dim3(DIM / 256, M / 32), 128, 0, stream>>>(xb, wqb, qf, DIM, DIM, DIM, DIM);
  gemm_bf16_nt_tdm<<<dim3(KVD / 256, M / 32), 128, 0, stream>>>(xb, wkb, kf, DIM, DIM, DIM, KVD);
  gemm_bf16_nt_tdm<<<dim3(KVD / 256, M / 32), 128, 0, stream>>>(xb, wvb, vf, DIM, DIM, DIM, KVD);
  // 3) RMSNorm + RoPE (+gain and 1/sqrt(D) folded into Q); layouts for attention
  rmsnorm_rope<<<Bn * SEQ * NH,  32, 0, stream>>>(qf, qb, qg,      NH,  0.08838834764831843f);
  rmsnorm_rope<<<Bn * SEQ * NKV, 32, 0, stream>>>(kf, kb, nullptr, NKV, 1.0f);
  pack_v_t<<<PG, PB, 0, stream>>>(vf, vb, Bn);
  // 4) causal flash attention
  flash_attn_wmma<<<dim3(SEQ / 64, NH, Bn), 128, 0, stream>>>(qb, kb, vb, yf);
  // 5) output projection
  pack_bf16<<<PG, PB, 0, stream>>>(yf, yb, M * DIM);
  gemm_bf16_nt_tdm<<<dim3(DIM / 256, M / 32), 128, 0, stream>>>(yb, wob, out, DIM, DIM, DIM, DIM);
}
